// AFSLSTM_91268055040200
// MI455X (gfx1250) — compile-verified, matrix operand order checked by reading, not run on hardware
//
#include <hip/hip_runtime.h>
#include <hip/hip_bf16.h>

// ---------------------------------------------------------------------------
// Problem constants (match reference)
// ---------------------------------------------------------------------------
#define Bc 8192
#define Fc 2048
#define Hc 1024

typedef __attribute__((ext_vector_type(8)))  __bf16 v8bf;
typedef __attribute__((ext_vector_type(16))) __bf16 v16bf;
typedef __attribute__((ext_vector_type(8)))  float  v8f;
typedef __attribute__((ext_vector_type(4)))  unsigned int u32x4;
typedef __attribute__((ext_vector_type(8)))  int i32x8;
typedef __attribute__((ext_vector_type(4)))  int i32x4;

union FragAB { v16bf v; v8bf h[2]; };

__device__ __forceinline__ float sigmoidf_(float v) {
    return 1.0f / (1.0f + __expf(-v));
}

// ---------------------------------------------------------------------------
// TDM: DMA one 128-row x 64-elem bf16 tile (row stride K elems) into LDS.
// LDS layout: row r at ldsAddr + r*144 bytes (128B data + 16B pad from the
// D# pad feature -> ds_load_b128 fragment reads are bank-conflict free).
// D# bit layout per CDNA5 ISA.
// ---------------------------------------------------------------------------
__device__ __forceinline__ void tdm_load_tile(const __bf16* gbase, unsigned ldsAddr, int K)
{
    const unsigned long long ga = (unsigned long long)gbase;
    u32x4 g0;
    g0[0] = 1u;                                           // count=1, user desc
    g0[1] = ldsAddr;                                      // lds_addr
    g0[2] = (unsigned)(ga & 0xffffffffull);               // global_addr[31:0]
    g0[3] = (unsigned)((ga >> 32) & 0x1ffffffull)         // global_addr[56:32]
          | 0x80000000u;                                  // type=2 ("image")
    const unsigned td0 = (unsigned)K;                     // tensor_dim0
    const unsigned td1 = 1u << 20;                        // tensor_dim1 (no OOB)
    i32x8 g1;
    g1[0] = (int)((1u << 16)        // data_size = 2 bytes
                | (1u << 20)        // pad_enable
                | (4u << 22)        // pad_interval: 32 dwords (=128B row)
                | (3u << 25));      // pad_amount: 4 dwords (=16B)
    g1[1] = (int)((td0 & 0xffffu) << 16);                 // tensor_dim0[15:0]
    g1[2] = (int)((td0 >> 16) | ((td1 & 0xffffu) << 16)); // dim0 hi | dim1 lo
    g1[3] = (int)((td1 >> 16) | (64u << 16));             // dim1 hi | tile_dim0=64
    g1[4] = 128;                                          // tile_dim1=128, tile_dim2=0
    g1[5] = K;                                            // tensor_dim0_stride[31:0]
    g1[6] = 0;                                            // stride hi, dim1_stride lo
    g1[7] = 0;
    const i32x4 gz = {0, 0, 0, 0};
#if defined(__clang_major__) && __clang_major__ >= 23
    const i32x8 gz8 = {0, 0, 0, 0, 0, 0, 0, 0};
    __builtin_amdgcn_tensor_load_to_lds(g0, g1, gz, gz, gz8, 0);
#else
    __builtin_amdgcn_tensor_load_to_lds(g0, g1, gz, gz, 0);
#endif
}

// ---------------------------------------------------------------------------
// TDM + LDS double-buffered bf16 WMMA GEMM:
//   C[M,N] = A[M,K] * Wt[N,K]^T + bias[N]      (row-major, K multiple of 64)
// Block = 256 threads = 8 waves (4 along M x 2 along N), block tile 128x128,
// wave tile 32x64. Wave 0 drives the TDM; s_wait_tensorcnt + barrier syncs.
// MODE 0: store f32   MODE 1: relu->f32   MODE 2: bf16 aux*sigmoid(c)
// ---------------------------------------------------------------------------
template <int MODE>
__global__ __launch_bounds__(256)
void gemm_bf16_wmma_tdm(const __bf16* __restrict__ A,
                        const __bf16* __restrict__ Wt,
                        const float*  __restrict__ bias,
                        float*        __restrict__ Cf,
                        __bf16*       __restrict__ Cb,
                        const float*  __restrict__ aux,
                        int M, int N, int K)
{
    // 4 buffers of 128 rows * 144B: A0,B0,A1,B1
    __shared__ __align__(16) char smem[4 * 128 * 144];
    const unsigned ldsBase = (unsigned)(size_t)&smem[0]; // low 32 bits = LDS offset

    const int lane = threadIdx.x & 31;
    const int wave = threadIdx.x >> 5;
    const int wm   = wave & 3;                  // wave row (4 x 32 rows)
    const int wn   = wave >> 2;                 // wave col (2 x 64 cols)
    const int m0   = blockIdx.y * 128;
    const int n0   = blockIdx.x * 128;
    const int r    = lane & 15;
    const int half = lane >> 4;

    const unsigned SZ = 128u * 144u;            // 18432 bytes per tile buffer

    if (wave == 0) {
        tdm_load_tile(A  + (size_t)m0 * K, ldsBase + 0u * SZ, K);
        tdm_load_tile(Wt + (size_t)n0 * K, ldsBase + 1u * SZ, K);
    }

    v8f acc[2][4];
#pragma unroll
    for (int mi = 0; mi < 2; ++mi)
#pragma unroll
        for (int ni = 0; ni < 4; ++ni)
            acc[mi][ni] = (v8f){0.f, 0.f, 0.f, 0.f, 0.f, 0.f, 0.f, 0.f};

    const int nk = K >> 6;                      // K-steps of 64
    for (int kt = 0; kt < nk; ++kt) {
        const int cur = kt & 1;
        if (wave == 0)
            __builtin_amdgcn_s_wait_tensorcnt(0);   // current pair landed
        __syncthreads();                            // publish LDS to all waves
        if (wave == 0 && (kt + 1) < nk) {           // overlap next DMA w/ math
            const unsigned nxt = (unsigned)(cur ^ 1) * 2u * SZ;
            tdm_load_tile(A  + (size_t)m0 * K + ((kt + 1) << 6), ldsBase + nxt, K);
            tdm_load_tile(Wt + (size_t)n0 * K + ((kt + 1) << 6), ldsBase + nxt + SZ, K);
        }

        const char* sA = smem + (unsigned)cur * 2u * SZ;
        const char* sB = sA + SZ;
#pragma unroll
        for (int kk = 0; kk < 2; ++kk) {
            const int kb = kk * 64 + half * 16;     // byte offset inside row
            FragAB a[2], b[4];
#pragma unroll
            for (int mi = 0; mi < 2; ++mi) {
                const char* p = sA + (wm * 32 + mi * 16 + r) * 144 + kb;
                a[mi].h[0] = *(const v8bf*)(p);
                a[mi].h[1] = *(const v8bf*)(p + 32);
            }
#pragma unroll
            for (int ni = 0; ni < 4; ++ni) {
                const char* p = sB + (wn * 64 + ni * 16 + r) * 144 + kb;
                b[ni].h[0] = *(const v8bf*)(p);
                b[ni].h[1] = *(const v8bf*)(p + 32);
            }
#pragma unroll
            for (int mi = 0; mi < 2; ++mi)
#pragma unroll
                for (int ni = 0; ni < 4; ++ni)
                    acc[mi][ni] = __builtin_amdgcn_wmma_f32_16x16x32_bf16(
                        false, a[mi].v, false, b[ni].v, (short)0, acc[mi][ni],
                        false, false);
        }
        __syncthreads();                            // done reading cur buffer
    }

    // Epilogue: element rr of each v8f -> row (rr + 8*half), col (lane&15).
#pragma unroll
    for (int ni = 0; ni < 4; ++ni) {
        const int col = n0 + wn * 64 + ni * 16 + r;
        const float bv = bias[col];
#pragma unroll
        for (int mi = 0; mi < 2; ++mi) {
#pragma unroll
            for (int rr = 0; rr < 8; ++rr) {
                const int row = m0 + wm * 32 + mi * 16 + half * 8 + rr;
                const size_t idx = (size_t)row * N + col;
                float v = acc[mi][ni][rr] + bv;
                if (MODE == 0) {
                    Cf[idx] = v;
                } else if (MODE == 1) {
                    Cf[idx] = v > 0.f ? v : 0.f;
                } else {
                    Cb[idx] = (__bf16)(aux[idx] * sigmoidf_(v));
                }
            }
        }
    }
}

// ---------------------------------------------------------------------------
// Direct-load WMMA GEMM for the small head layer (N=64): 8 waves stacked in M,
// block tile 256x64, wave tile 32x64. Relu epilogue, f32 out.
// ---------------------------------------------------------------------------
__global__ __launch_bounds__(256)
void gemm_bf16_wmma_n64(const __bf16* __restrict__ A,
                        const __bf16* __restrict__ Wt,
                        const float*  __restrict__ bias,
                        float*        __restrict__ Cf,
                        int M, int N, int K)
{
    const int lane = threadIdx.x & 31;
    const int wave = threadIdx.x >> 5;
    const int m0   = blockIdx.y * 256 + wave * 32;
    const int r    = lane & 15;
    const int half = lane >> 4;

    const __bf16* __restrict__ a0p = A + (size_t)(m0 + r) * K + half * 8;
    const __bf16* __restrict__ a1p = a0p + (size_t)16 * K;
    const __bf16* __restrict__ bp0 = Wt + (size_t)( 0 + r) * K + half * 8;
    const __bf16* __restrict__ bp1 = Wt + (size_t)(16 + r) * K + half * 8;
    const __bf16* __restrict__ bp2 = Wt + (size_t)(32 + r) * K + half * 8;
    const __bf16* __restrict__ bp3 = Wt + (size_t)(48 + r) * K + half * 8;

    v8f acc[2][4];
#pragma unroll
    for (int mi = 0; mi < 2; ++mi)
#pragma unroll
        for (int ni = 0; ni < 4; ++ni)
            acc[mi][ni] = (v8f){0.f, 0.f, 0.f, 0.f, 0.f, 0.f, 0.f, 0.f};

    for (int k = 0; k < K; k += 32) {
        __builtin_prefetch(a0p + k + 32, 0, 3);
        __builtin_prefetch(bp0 + k + 32, 0, 3);

        FragAB a0, a1, b0, b1, b2, b3;
        a0.h[0] = *(const v8bf*)(a0p + k);
        a0.h[1] = *(const v8bf*)(a0p + k + 16);
        a1.h[0] = *(const v8bf*)(a1p + k);
        a1.h[1] = *(const v8bf*)(a1p + k + 16);
        b0.h[0] = *(const v8bf*)(bp0 + k);
        b0.h[1] = *(const v8bf*)(bp0 + k + 16);
        b1.h[0] = *(const v8bf*)(bp1 + k);
        b1.h[1] = *(const v8bf*)(bp1 + k + 16);
        b2.h[0] = *(const v8bf*)(bp2 + k);
        b2.h[1] = *(const v8bf*)(bp2 + k + 16);
        b3.h[0] = *(const v8bf*)(bp3 + k);
        b3.h[1] = *(const v8bf*)(bp3 + k + 16);

        acc[0][0] = __builtin_amdgcn_wmma_f32_16x16x32_bf16(false, a0.v, false, b0.v, (short)0, acc[0][0], false, false);
        acc[0][1] = __builtin_amdgcn_wmma_f32_16x16x32_bf16(false, a0.v, false, b1.v, (short)0, acc[0][1], false, false);
        acc[0][2] = __builtin_amdgcn_wmma_f32_16x16x32_bf16(false, a0.v, false, b2.v, (short)0, acc[0][2], false, false);
        acc[0][3] = __builtin_amdgcn_wmma_f32_16x16x32_bf16(false, a0.v, false, b3.v, (short)0, acc[0][3], false, false);
        acc[1][0] = __builtin_amdgcn_wmma_f32_16x16x32_bf16(false, a1.v, false, b0.v, (short)0, acc[1][0], false, false);
        acc[1][1] = __builtin_amdgcn_wmma_f32_16x16x32_bf16(false, a1.v, false, b1.v, (short)0, acc[1][1], false, false);
        acc[1][2] = __builtin_amdgcn_wmma_f32_16x16x32_bf16(false, a1.v, false, b2.v, (short)0, acc[1][2], false, false);
        acc[1][3] = __builtin_amdgcn_wmma_f32_16x16x32_bf16(false, a1.v, false, b3.v, (short)0, acc[1][3], false, false);
    }

#pragma unroll
    for (int ni = 0; ni < 4; ++ni) {
        const int col = ni * 16 + r;
        const float bv = bias[col];
#pragma unroll
        for (int mi = 0; mi < 2; ++mi) {
#pragma unroll
            for (int rr = 0; rr < 8; ++rr) {
                const int row = m0 + mi * 16 + half * 8 + rr;
                const float v = acc[mi][ni][rr] + bv;
                Cf[(size_t)row * N + col] = v > 0.f ? v : 0.f;
            }
        }
    }
}

// ---------------------------------------------------------------------------
// LayerNorm over F=2048 per row, affine, output bf16
// ---------------------------------------------------------------------------
__global__ __launch_bounds__(256)
void layernorm_to_bf16(const float* __restrict__ x,
                       const float* __restrict__ g,
                       const float* __restrict__ b,
                       __bf16* __restrict__ xn)
{
    __shared__ float s1[256];
    __shared__ float s2[256];
    const int row = blockIdx.x;
    const int t   = threadIdx.x;
    const float* xr = x + (size_t)row * Fc;

    float s = 0.f, ss = 0.f;
    for (int i = t; i < Fc; i += 256) {
        float v = xr[i];
        s += v; ss += v * v;
    }
    s1[t] = s; s2[t] = ss;
    __syncthreads();
    for (int o = 128; o > 0; o >>= 1) {
        if (t < o) { s1[t] += s1[t + o]; s2[t] += s2[t + o]; }
        __syncthreads();
    }
    const float mu  = s1[0] * (1.0f / Fc);
    const float var = s2[0] * (1.0f / Fc) - mu * mu;
    const float inv = rsqrtf(var + 1e-5f);

    __bf16* orow = xn + (size_t)row * Fc;
    for (int i = t; i < Fc; i += 256)
        orow[i] = (__bf16)((xr[i] - mu) * inv * g[i] + b[i]);
}

// ---------------------------------------------------------------------------
// LSTM single-step activation: h = sig(o)*tanh(sig(i)*tanh(g)); f unused
// ---------------------------------------------------------------------------
__global__ __launch_bounds__(256)
void lstm_act(const float* __restrict__ gates, __bf16* __restrict__ feat, int dirOff)
{
    const int idx = blockIdx.x * 256 + threadIdx.x;   // over B*H
    const int bi = idx >> 10;                         // / H
    const int h  = idx & (Hc - 1);                    // % H
    const float* gr = gates + (size_t)bi * (4 * Hc);
    const float gi = gr[h];
    const float gg = gr[2 * Hc + h];
    const float go = gr[3 * Hc + h];
    const float c  = sigmoidf_(gi) * tanhf(gg);
    const float hv = sigmoidf_(go) * tanhf(c);
    feat[(size_t)bi * (2 * Hc) + dirOff + h] = (__bf16)hv;
}

// ---------------------------------------------------------------------------
// Helpers: f32->bf16 convert, bias sum, final head dot
// ---------------------------------------------------------------------------
__global__ __launch_bounds__(256)
void f32_to_bf16(const float* __restrict__ in, __bf16* __restrict__ out, int n)
{
    const int i = blockIdx.x * 256 + threadIdx.x;
    if (i < n) out[i] = (__bf16)in[i];
}

__global__ __launch_bounds__(256)
void bias_sum(const float* __restrict__ a, const float* __restrict__ b,
              float* __restrict__ o, int n)
{
    const int i = blockIdx.x * 256 + threadIdx.x;
    if (i < n) o[i] = a[i] + b[i];
}

__global__ __launch_bounds__(256)
void head_kernel(const float* __restrict__ hid, const float* __restrict__ W2,
                 const float* __restrict__ b2, float* __restrict__ out)
{
    const int bi = blockIdx.x * 256 + threadIdx.x;
    if (bi >= Bc) return;
    const float* hr = hid + (size_t)bi * 64;
    float s = 0.f;
#pragma unroll
    for (int j = 0; j < 64; ++j) s += hr[j] * W2[j];
    out[bi] = s + b2[0];
}

// ---------------------------------------------------------------------------
// Orchestration
// ---------------------------------------------------------------------------
extern "C" void kernel_launch(void* const* d_in, const int* in_sizes, int n_in,
                              void* d_out, int out_size, void* d_ws, size_t ws_size,
                              hipStream_t stream)
{
    (void)in_sizes; (void)n_in; (void)out_size; (void)ws_size;

    const float* x      = (const float*)d_in[0];
    const float* ln_g   = (const float*)d_in[1];
    const float* ln_b   = (const float*)d_in[2];
    const float* Wg     = (const float*)d_in[3];
    const float* bg     = (const float*)d_in[4];
    const float* W_ih_f = (const float*)d_in[5];
    const float* b_ih_f = (const float*)d_in[7];
    const float* b_hh_f = (const float*)d_in[8];
    const float* W_ih_b = (const float*)d_in[9];
    const float* b_ih_b = (const float*)d_in[11];
    const float* b_hh_b = (const float*)d_in[12];
    const float* W1     = (const float*)d_in[13];
    const float* b1     = (const float*)d_in[14];
    const float* W2     = (const float*)d_in[15];
    const float* b2     = (const float*)d_in[16];
    float* out = (float*)d_out;

    // Workspace layout (bytes); feat reuses xn, hid reuses Wg_bf16.
    char* ws = (char*)d_ws;
    const size_t MBy = 1ull << 20;
    __bf16* xn_bf   = (__bf16*)(ws + 0 * MBy);          // 32 MB, later feat
    __bf16* xg_bf   = (__bf16*)(ws + 32 * MBy);         // 32 MB
    __bf16* wg_bf   = (__bf16*)(ws + 64 * MBy);         // 8 MB, later hid
    __bf16* wf_bf   = (__bf16*)(ws + 72 * MBy);         // 16 MB
    __bf16* wb_bf   = (__bf16*)(ws + 88 * MBy);         // 16 MB
    __bf16* w1_bf   = (__bf16*)(ws + 104 * MBy);        // 0.25 MB
    float*  bsum_f  = (float*)(ws + 105 * MBy);         // 16 KB
    float*  bsum_b  = (float*)(ws + 105 * MBy + 65536); // 16 KB
    float*  gates   = (float*)(ws + 106 * MBy);         // 128 MB
    __bf16* feat_bf = xn_bf;
    float*  hid     = (float*)wg_bf;

    // 1) weight conversions + combined LSTM biases
    f32_to_bf16<<<(Fc * Fc) / 256, 256, 0, stream>>>(Wg, wg_bf, Fc * Fc);
    f32_to_bf16<<<(4 * Hc * Fc) / 256, 256, 0, stream>>>(W_ih_f, wf_bf, 4 * Hc * Fc);
    f32_to_bf16<<<(4 * Hc * Fc) / 256, 256, 0, stream>>>(W_ih_b, wb_bf, 4 * Hc * Fc);
    f32_to_bf16<<<(64 * 2 * Hc) / 256, 256, 0, stream>>>(W1, w1_bf, 64 * 2 * Hc);
    bias_sum<<<(4 * Hc) / 256, 256, 0, stream>>>(b_ih_f, b_hh_f, bsum_f, 4 * Hc);
    bias_sum<<<(4 * Hc) / 256, 256, 0, stream>>>(b_ih_b, b_hh_b, bsum_b, 4 * Hc);

    // 2) LayerNorm -> bf16
    layernorm_to_bf16<<<Bc, 256, 0, stream>>>(x, ln_g, ln_b, xn_bf);

    // 3) FeatureGate GEMM (TDM-staged) with fused sigmoid*x epilogue -> xg bf16
    gemm_bf16_wmma_tdm<2><<<dim3(Fc / 128, Bc / 128), 256, 0, stream>>>(
        xn_bf, wg_bf, bg, nullptr, xg_bf, x, Bc, Fc, Fc);

    // 4) LSTM forward gates GEMM, then activation into feat[:, :H]
    gemm_bf16_wmma_tdm<0><<<dim3((4 * Hc) / 128, Bc / 128), 256, 0, stream>>>(
        xg_bf, wf_bf, bsum_f, gates, nullptr, nullptr, Bc, 4 * Hc, Fc);
    lstm_act<<<(Bc * Hc) / 256, 256, 0, stream>>>(gates, feat_bf, 0);

    // 5) LSTM backward gates GEMM (reuses gates buffer), activation -> feat[:, H:]
    gemm_bf16_wmma_tdm<0><<<dim3((4 * Hc) / 128, Bc / 128), 256, 0, stream>>>(
        xg_bf, wb_bf, bsum_b, gates, nullptr, nullptr, Bc, 4 * Hc, Fc);
    lstm_act<<<(Bc * Hc) / 256, 256, 0, stream>>>(gates, feat_bf, Hc);

    // 6) head hidden layer GEMM with fused ReLU (N=64)
    gemm_bf16_wmma_n64<<<dim3(1, Bc / 256), 256, 0, stream>>>(
        feat_bf, w1_bf, b1, hid, Bc, 64, 2 * Hc);

    // 7) final 64->1 dot + bias
    head_kernel<<<Bc / 256, 256, 0, stream>>>(hid, W2, b2, out);
}
